// Net_21947282882692
// MI455X (gfx1250) — compile-verified, matrix operand order checked by reading, not run on hardware
//
#include <hip/hip_runtime.h>

typedef _Float16 v16h __attribute__((ext_vector_type(16)));
typedef float    v8f  __attribute__((ext_vector_type(8)));

#define WMMA(a, b, c) __builtin_amdgcn_wmma_f32_16x16x32_f16(false, (a), false, (b), (short)0, (c), false, false)

// ---------------------------------------------------------------------------
// CDNA5 async global->LDS copy (ASYNCcnt path), with portable fallback.
// Builtin signature (from hipcc diagnostic): (v4i AS1*, v4i AS3*, imm, imm).
// ---------------------------------------------------------------------------
#if __has_builtin(__builtin_amdgcn_global_load_async_to_lds_b128) && \
    __has_builtin(__builtin_amdgcn_s_wait_asynccnt)
#define USE_ASYNC_LDS 1
#else
#define USE_ASYNC_LDS 0
#endif

typedef int v4i_vs __attribute__((vector_size(16)));

__device__ __forceinline__ void async_copy16(void* lds_dst, const void* gsrc) {
#if USE_ASYNC_LDS
    __builtin_amdgcn_global_load_async_to_lds_b128(
        (__attribute__((address_space(1))) v4i_vs*)(gsrc),
        (__attribute__((address_space(3))) v4i_vs*)(lds_dst), 0, 0);
#else
    *(uint4*)lds_dst = *(const uint4*)gsrc;
#endif
}

__device__ __forceinline__ void async_wait_all() {
#if USE_ASYNC_LDS
    __builtin_amdgcn_s_wait_asynccnt(0);
#endif
}

// ---------------------------------------------------------------------------
// WMMA tile loaders (ISA 7.12.2 16-bit A layout; B mirrored with N in lanes).
// A tile: [16 x 32] from row-major src (ld elements), K offset k0.
// Works from LDS (f16/f32) or directly from global f32.
// ---------------------------------------------------------------------------
template <typename T>
__device__ __forceinline__ v16h ld_mat_A(const T* src, int ld, int k0, int lane) {
    int m = lane & 15, kh = lane >> 4;
    v16h r;
#pragma unroll
    for (int v = 0; v < 8; ++v) {
        int k = k0 + ((v >> 2) << 4) + (kh << 3) + ((v & 3) << 1);
        r[2 * v]     = (_Float16)src[m * ld + k];
        r[2 * v + 1] = (_Float16)src[m * ld + k + 1];
    }
    return r;
}

// B(k,n) = W[n*ld + k] for row-major weight W[N,K]; tile at (n0, k0).
template <typename T>
__device__ __forceinline__ v16h ld_mat_B(const T* w, int ld, int n0, int k0, int lane) {
    int n = lane & 15, kh = lane >> 4;
    v16h r;
#pragma unroll
    for (int v = 0; v < 8; ++v) {
        int k = k0 + ((v >> 2) << 4) + (kh << 3) + ((v & 3) << 1);
        r[2 * v]     = (_Float16)w[(n0 + n) * ld + k];
        r[2 * v + 1] = (_Float16)w[(n0 + n) * ld + k + 1];
    }
    return r;
}

__device__ __forceinline__ void cp_f16(_Float16* dst, const float* src, int n, int tid, int nt) {
    for (int i = tid; i < n; i += nt) dst[i] = (_Float16)src[i];
}
__device__ __forceinline__ void cp_f32(float* dst, const float* src, int n, int tid, int nt) {
    for (int i = tid; i < n; i += nt) dst[i] = src[i];
}
__device__ __forceinline__ float sigf(float x) { return 1.0f / (1.0f + expf(-x)); }

// ===========================================================================
// K1: state tower.  core_in[2048,64] = [relu MLP(state_table) || relu(state_other@W_s0)]
// A-tiles stream straight from global f32 (single use, no LDS staging).
// ===========================================================================
__global__ __launch_bounds__(256) void k_state_tower(
    const float* __restrict__ state_table, const float* __restrict__ state_other,
    const float* __restrict__ W_t0, const float* __restrict__ b_t0,
    const float* __restrict__ W_t1, const float* __restrict__ b_t1,
    const float* __restrict__ W_s0, const float* __restrict__ b_s0,
    _Float16* __restrict__ core_in) {
    __shared__ _Float16 w0[128 * 128];
    __shared__ _Float16 w1[32 * 128];
    __shared__ _Float16 ws0[32 * 64];
    __shared__ float bb0[128], bb1[32], bbs[32];
    __shared__ alignas(16) _Float16 bufM[8][16 * 128];

    int tid = threadIdx.x;
    cp_f16(w0, W_t0, 128 * 128, tid, 256);
    cp_f16(w1, W_t1, 32 * 128, tid, 256);
    cp_f16(ws0, W_s0, 32 * 64, tid, 256);
    cp_f32(bb0, b_t0, 128, tid, 256);
    cp_f32(bb1, b_t1, 32, tid, 256);
    cp_f32(bbs, b_s0, 32, tid, 256);
    __syncthreads();

    int wave = tid >> 5, lane = tid & 31;
    int row0 = blockIdx.x * 128 + wave * 16;
    const float* A0 = state_table + (long)row0 * 128;

    v8f z = {};
    v8f acc[8];
#pragma unroll
    for (int q = 0; q < 8; ++q) acc[q] = z;
#pragma unroll
    for (int ks = 0; ks < 4; ++ks) {
        v16h a = ld_mat_A(A0, 128, ks * 32, lane);
#pragma unroll
        for (int q = 0; q < 8; ++q) {
            v16h b = ld_mat_B(w0, 128, q * 16, ks * 32, lane);
            acc[q] = WMMA(a, b, acc[q]);
        }
    }
    int n_ = lane & 15, mh = lane >> 4;
#pragma unroll
    for (int q = 0; q < 8; ++q)
#pragma unroll
        for (int r = 0; r < 8; ++r) {
            int n = q * 16 + n_, m = r + 8 * mh;
            bufM[wave][m * 128 + n] = (_Float16)fmaxf(acc[q][r] + bb0[n], 0.0f);
        }
    __syncthreads();

    v8f acc2[2] = {z, z};
#pragma unroll
    for (int ks = 0; ks < 4; ++ks) {
        v16h a = ld_mat_A(bufM[wave], 128, ks * 32, lane);
#pragma unroll
        for (int q = 0; q < 2; ++q) {
            v16h b = ld_mat_B(w1, 128, q * 16, ks * 32, lane);
            acc2[q] = WMMA(a, b, acc2[q]);
        }
    }
#pragma unroll
    for (int q = 0; q < 2; ++q)
#pragma unroll
        for (int r = 0; r < 8; ++r) {
            int n = q * 16 + n_, m = r + 8 * mh;
            core_in[(row0 + m) * 64 + n] = (_Float16)fmaxf(acc2[q][r] + bb1[n], 0.0f);
        }

    // state_other branch: [16,64] @ W_s0^T -> [16,32], A straight from global
    const float* A1 = state_other + (long)row0 * 64;
    v8f acc3[2] = {z, z};
#pragma unroll
    for (int ks = 0; ks < 2; ++ks) {
        v16h a = ld_mat_A(A1, 64, ks * 32, lane);
#pragma unroll
        for (int q = 0; q < 2; ++q) {
            v16h b = ld_mat_B(ws0, 64, q * 16, ks * 32, lane);
            acc3[q] = WMMA(a, b, acc3[q]);
        }
    }
#pragma unroll
    for (int q = 0; q < 2; ++q)
#pragma unroll
        for (int r = 0; r < 8; ++r) {
            int n = q * 16 + n_, m = r + 8 * mh;
            core_in[(row0 + m) * 64 + 32 + n] = (_Float16)fmaxf(acc3[q][r] + bbs[n], 0.0f);
        }
}

// ===========================================================================
// K2: persistent 2-layer done-masked LSTM, one workgroup, WMMA gates.
// ===========================================================================
__device__ __forceinline__ void lstm_gates(const _Float16* x, const _Float16* h,
                                           const float* Wih, const float* Whh,
                                           const float* bsum, float* gates,
                                           int wave, int lane) {
    for (int rep = 0; rep < 4; ++rep) {
        int idx = wave + rep * 8;           // 32 output tiles: 2 M-tiles x 16 N-tiles
        int mt = idx >> 4, ntile = idx & 15;
        v8f acc = {};
#pragma unroll
        for (int ks = 0; ks < 2; ++ks) {
            v16h a = ld_mat_A(x + mt * 16 * 64, 64, ks * 32, lane);
            v16h b = ld_mat_B(Wih, 64, ntile * 16, ks * 32, lane);
            acc = WMMA(a, b, acc);
            v16h a2 = ld_mat_A(h + mt * 16 * 64, 64, ks * 32, lane);
            v16h b2 = ld_mat_B(Whh, 64, ntile * 16, ks * 32, lane);
            acc = WMMA(a2, b2, acc);
        }
        int n_ = lane & 15, mh = lane >> 4;
#pragma unroll
        for (int r = 0; r < 8; ++r) {
            int n = ntile * 16 + n_, m = mt * 16 + r + 8 * mh;
            gates[m * 256 + n] = acc[r] + bsum[n];
        }
    }
}

__global__ __launch_bounds__(256) void k_lstm(
    const _Float16* __restrict__ core_in,
    const float* __restrict__ Wih0, const float* __restrict__ Whh0,
    const float* __restrict__ bih0, const float* __restrict__ bhh0,
    const float* __restrict__ Wih1, const float* __restrict__ Whh1,
    const float* __restrict__ bih1, const float* __restrict__ bhh1,
    const float* __restrict__ h0, const float* __restrict__ c0,
    const unsigned char* __restrict__ done,
    float* __restrict__ state, float* __restrict__ hT, float* __restrict__ cT) {
    __shared__ _Float16 h_l[2 * 2048];   // [layer][b*64+j]
    __shared__ float c_l[2 * 2048];
    __shared__ float gates[32 * 256];
    __shared__ float bs0[256], bs1[256];

    int tid = threadIdx.x;
    for (int i = tid; i < 256; i += 256) { bs0[i] = bih0[i] + bhh0[i]; bs1[i] = bih1[i] + bhh1[i]; }
    for (int i = tid; i < 4096; i += 256) { h_l[i] = (_Float16)h0[i]; c_l[i] = c0[i]; }
    __syncthreads();

    int wave = tid >> 5, lane = tid & 31;
    for (int t = 0; t < 64; ++t) {
        for (int i = tid; i < 4096; i += 256) {
            int b = (i >> 6) & 31;
            float nd = done[t * 32 + b] ? 0.0f : 1.0f;
            h_l[i] = (_Float16)((float)h_l[i] * nd);
            c_l[i] *= nd;
        }
        __syncthreads();
        lstm_gates(core_in + (long)t * 2048, h_l, Wih0, Whh0, bs0, gates, wave, lane);
        __syncthreads();
        for (int i = tid; i < 2048; i += 256) {
            int b = i >> 6, j = i & 63;
            float gi = gates[b * 256 + j], gf = gates[b * 256 + 64 + j];
            float gg = gates[b * 256 + 128 + j], go = gates[b * 256 + 192 + j];
            float cc = sigf(gf) * c_l[i] + sigf(gi) * tanhf(gg);
            c_l[i] = cc;
            h_l[i] = (_Float16)(sigf(go) * tanhf(cc));
        }
        __syncthreads();
        lstm_gates(h_l, h_l + 2048, Wih1, Whh1, bs1, gates, wave, lane);
        __syncthreads();
        for (int i = tid; i < 2048; i += 256) {
            int b = i >> 6, j = i & 63;
            float gi = gates[b * 256 + j], gf = gates[b * 256 + 64 + j];
            float gg = gates[b * 256 + 128 + j], go = gates[b * 256 + 192 + j];
            float cc = sigf(gf) * c_l[2048 + i] + sigf(gi) * tanhf(gg);
            float hh = sigf(go) * tanhf(cc);
            c_l[2048 + i] = cc;
            h_l[2048 + i] = (_Float16)hh;
            state[((long)t * 32 + b) * 64 + j] = hh;
        }
        __syncthreads();
    }
    for (int i = tid; i < 4096; i += 256) { hT[i] = (float)h_l[i]; cT[i] = c_l[i]; }
}

// ===========================================================================
// K3: ragged actions tower feats + uniform-K=4 segment sum -> actions[:,0:32]
// Weights land in LDS via CDNA5 async global->LDS b128 (ASYNCcnt);
// A-tiles stream directly from global (no staging, single-use data).
// ===========================================================================
__global__ __launch_bounds__(256) void k_feats(
    const float* __restrict__ actions_table,
    const float* __restrict__ W_t0, const float* __restrict__ b_t0,
    const float* __restrict__ W_t1, const float* __restrict__ b_t1,
    float* __restrict__ actions) {
    __shared__ alignas(16) float w0f[128 * 128];   // 64 KB, raw async copy
    __shared__ alignas(16) float w1f[32 * 128];    // 16 KB
    __shared__ float bb0[128], bb1[32];
    __shared__ alignas(16) _Float16 bufM[8][16 * 128];

    int tid = threadIdx.x;
    for (int i = tid * 4; i < 128 * 128; i += 256 * 4) async_copy16(&w0f[i], &W_t0[i]);
    for (int i = tid * 4; i < 32 * 128; i += 256 * 4) async_copy16(&w1f[i], &W_t1[i]);
    cp_f32(bb0, b_t0, 128, tid, 256);
    cp_f32(bb1, b_t1, 32, tid, 256);
    async_wait_all();
    __syncthreads();

    int wave = tid >> 5, lane = tid & 31;
    long row0 = (long)blockIdx.x * 128 + wave * 16;
    const float* A0 = actions_table + row0 * 128;
    __builtin_prefetch(actions_table + (row0 + 128) * 128 + lane * 32, 0, 1);

    v8f z = {};
    v8f acc[8];
#pragma unroll
    for (int q = 0; q < 8; ++q) acc[q] = z;
#pragma unroll
    for (int ks = 0; ks < 4; ++ks) {
        v16h a = ld_mat_A(A0, 128, ks * 32, lane);
#pragma unroll
        for (int q = 0; q < 8; ++q) {
            v16h b = ld_mat_B(w0f, 128, q * 16, ks * 32, lane);
            acc[q] = WMMA(a, b, acc[q]);
        }
    }
    int n_ = lane & 15, mh = lane >> 4;
#pragma unroll
    for (int q = 0; q < 8; ++q)
#pragma unroll
        for (int r = 0; r < 8; ++r) {
            int n = q * 16 + n_, m = r + 8 * mh;
            bufM[wave][m * 128 + n] = (_Float16)fmaxf(acc[q][r] + bb0[n], 0.0f);
        }
    __syncthreads();

    v8f acc2[2] = {z, z};
#pragma unroll
    for (int ks = 0; ks < 4; ++ks) {
        v16h a = ld_mat_A(bufM[wave], 128, ks * 32, lane);
#pragma unroll
        for (int q = 0; q < 2; ++q) {
            v16h b = ld_mat_B(w1f, 128, q * 16, ks * 32, lane);
            acc2[q] = WMMA(a, b, acc2[q]);
        }
    }
    // relu(feats+bias) then segment-sum rows 4-at-a-time directly from the
    // WMMA C layout: VGPRs 0..3 hold rows M=0..3 (lanes<16) / M=8..11 (lanes>=16).
    long segbase = row0 >> 2;
#pragma unroll
    for (int q = 0; q < 2; ++q) {
        int n = q * 16 + n_;
        float fr[8];
#pragma unroll
        for (int r = 0; r < 8; ++r) fr[r] = fmaxf(acc2[q][r] + bb1[n], 0.0f);
        float sA = fr[0] + fr[1] + fr[2] + fr[3];
        float sB = fr[4] + fr[5] + fr[6] + fr[7];
        actions[(segbase + mh * 2 + 0) * 64 + n] = sA;
        actions[(segbase + mh * 2 + 1) * 64 + n] = sB;
    }
}

// ===========================================================================
// K4: actions_other tower (K padded 16->32) -> actions[:,32:64]
// ===========================================================================
__global__ __launch_bounds__(256) void k_ao(
    const float* __restrict__ actions_other,
    const float* __restrict__ W_a0, const float* __restrict__ b_a0,
    float* __restrict__ actions) {
    __shared__ _Float16 wB[32 * 32];
    __shared__ float bb[32];
    __shared__ alignas(16) _Float16 bufA[8][16 * 32];
    int tid = threadIdx.x;
    for (int i = tid; i < 32 * 32; i += 256) {
        int n = i >> 5, k = i & 31;
        wB[i] = (k < 16) ? (_Float16)W_a0[n * 16 + k] : (_Float16)0.0f;
    }
    cp_f32(bb, b_a0, 32, tid, 256);
    int wave = tid >> 5, lane = tid & 31;
    long s0 = (long)blockIdx.x * 128 + wave * 16;
    for (int i = lane; i < 16 * 32; i += 32) {
        int m = i >> 5, k = i & 31;
        bufA[wave][i] = (k < 16) ? (_Float16)actions_other[(s0 + m) * 16 + k] : (_Float16)0.0f;
    }
    __syncthreads();
    v8f z = {};
    v8f acc[2] = {z, z};
    v16h a = ld_mat_A(bufA[wave], 32, 0, lane);
#pragma unroll
    for (int q = 0; q < 2; ++q) {
        v16h b = ld_mat_B(wB, 32, q * 16, 0, lane);
        acc[q] = WMMA(a, b, acc[q]);
    }
    int n_ = lane & 15, mh = lane >> 4;
#pragma unroll
    for (int q = 0; q < 2; ++q)
#pragma unroll
        for (int r = 0; r < 8; ++r) {
            int n = q * 16 + n_, m = r + 8 * mh;
            actions[(s0 + m) * 64 + 32 + n] = fmaxf(acc[q][r] + bb[n], 0.0f);
        }
}

// ===========================================================================
// K5: prediction MLP [S,192]->128->128->64->1 -> logits
// ===========================================================================
__global__ __launch_bounds__(128) void k_pred(
    const float* __restrict__ state, const float* __restrict__ actions,
    const float* __restrict__ W_p0, const float* __restrict__ b_p0,
    const float* __restrict__ W_p1, const float* __restrict__ b_p1,
    const float* __restrict__ W_p2, const float* __restrict__ b_p2,
    const float* __restrict__ W_p3, const float* __restrict__ b_p3,
    float* __restrict__ logits) {
    __shared__ _Float16 w0[128 * 192];
    __shared__ _Float16 w1[128 * 128];
    __shared__ _Float16 w2[64 * 128];
    __shared__ float bb0[128], bb1[128], bb2[64], wp3[64];
    __shared__ alignas(16) _Float16 bufA[4][16 * 192];
    __shared__ alignas(16) _Float16 bufB[4][16 * 128];

    int tid = threadIdx.x;
    cp_f16(w0, W_p0, 128 * 192, tid, 128);
    cp_f16(w1, W_p1, 128 * 128, tid, 128);
    cp_f16(w2, W_p2, 64 * 128, tid, 128);
    cp_f32(bb0, b_p0, 128, tid, 128);
    cp_f32(bb1, b_p1, 128, tid, 128);
    cp_f32(bb2, b_p2, 64, tid, 128);
    cp_f32(wp3, W_p3, 64, tid, 128);

    int wave = tid >> 5, lane = tid & 31;
    long s0 = (long)blockIdx.x * 64 + wave * 16;
    long tb = s0 >> 6;   // 16-seg tiles never straddle a (t,b) group of 64

    for (int i = lane; i < 16 * 192; i += 32) {
        int m = i / 192, j = i % 192;
        float v;
        if (j < 64)       v = state[tb * 64 + j];
        else if (j < 128) v = actions[(s0 + m) * 64 + (j - 64)];
        else              v = state[tb * 64 + (j - 128)] * actions[(s0 + m) * 64 + (j - 128)];
        bufA[wave][i] = (_Float16)v;
    }
    __syncthreads();

    v8f z = {};
    v8f acc[8];
    int n_ = lane & 15, mh = lane >> 4;

    // layer p0: K=192, N=128
#pragma unroll
    for (int q = 0; q < 8; ++q) acc[q] = z;
#pragma unroll
    for (int ks = 0; ks < 6; ++ks) {
        v16h a = ld_mat_A(bufA[wave], 192, ks * 32, lane);
#pragma unroll
        for (int q = 0; q < 8; ++q) {
            v16h b = ld_mat_B(w0, 192, q * 16, ks * 32, lane);
            acc[q] = WMMA(a, b, acc[q]);
        }
    }
#pragma unroll
    for (int q = 0; q < 8; ++q)
#pragma unroll
        for (int r = 0; r < 8; ++r)
            bufB[wave][(r + 8 * mh) * 128 + q * 16 + n_] =
                (_Float16)fmaxf(acc[q][r] + bb0[q * 16 + n_], 0.0f);
    __syncthreads();

    // layer p1: K=128, N=128 (output overwrites bufA)
#pragma unroll
    for (int q = 0; q < 8; ++q) acc[q] = z;
#pragma unroll
    for (int ks = 0; ks < 4; ++ks) {
        v16h a = ld_mat_A(bufB[wave], 128, ks * 32, lane);
#pragma unroll
        for (int q = 0; q < 8; ++q) {
            v16h b = ld_mat_B(w1, 128, q * 16, ks * 32, lane);
            acc[q] = WMMA(a, b, acc[q]);
        }
    }
#pragma unroll
    for (int q = 0; q < 8; ++q)
#pragma unroll
        for (int r = 0; r < 8; ++r)
            bufA[wave][(r + 8 * mh) * 128 + q * 16 + n_] =
                (_Float16)fmaxf(acc[q][r] + bb1[q * 16 + n_], 0.0f);
    __syncthreads();

    // layer p2: K=128, N=64 (f32 output aliases bufB)
    v8f acc2[4] = {z, z, z, z};
#pragma unroll
    for (int ks = 0; ks < 4; ++ks) {
        v16h a = ld_mat_A(bufA[wave], 128, ks * 32, lane);
#pragma unroll
        for (int q = 0; q < 4; ++q) {
            v16h b = ld_mat_B(w2, 128, q * 16, ks * 32, lane);
            acc2[q] = WMMA(a, b, acc2[q]);
        }
    }
    float* bufC = (float*)&bufB[wave][0];
#pragma unroll
    for (int q = 0; q < 4; ++q)
#pragma unroll
        for (int r = 0; r < 8; ++r)
            bufC[(r + 8 * mh) * 64 + q * 16 + n_] =
                fmaxf(acc2[q][r] + bb2[q * 16 + n_], 0.0f);
    __syncthreads();

    // layer p3: N=1 dot
    if (lane < 16) {
        float s = b_p3[0];
        for (int j = 0; j < 64; ++j) s += bufC[lane * 64 + j] * wp3[j];
        logits[s0 + lane] = s;
    }
}

// ===========================================================================
// K6: baseline head [2048,64]->64->1 (A direct from global)
// ===========================================================================
__global__ __launch_bounds__(256) void k_baseline(
    const float* __restrict__ state,
    const float* __restrict__ W_b0, const float* __restrict__ b_b0,
    const float* __restrict__ W_b1, const float* __restrict__ b_b1,
    float* __restrict__ baseline) {
    __shared__ _Float16 w0[64 * 64];
    __shared__ float bb[64], wb1[64];
    __shared__ float bufC[8][16 * 64];
    int tid = threadIdx.x;
    cp_f16(w0, W_b0, 64 * 64, tid, 256);
    cp_f32(bb, b_b0, 64, tid, 256);
    cp_f32(wb1, W_b1, 64, tid, 256);
    __syncthreads();
    int wave = tid >> 5, lane = tid & 31;
    int row0 = blockIdx.x * 128 + wave * 16;
    const float* A0 = state + (long)row0 * 64;
    v8f z = {};
    v8f acc[4] = {z, z, z, z};
#pragma unroll
    for (int ks = 0; ks < 2; ++ks) {
        v16h a = ld_mat_A(A0, 64, ks * 32, lane);
#pragma unroll
        for (int q = 0; q < 4; ++q) {
            v16h b = ld_mat_B(w0, 64, q * 16, ks * 32, lane);
            acc[q] = WMMA(a, b, acc[q]);
        }
    }
    int n_ = lane & 15, mh = lane >> 4;
#pragma unroll
    for (int q = 0; q < 4; ++q)
#pragma unroll
        for (int r = 0; r < 8; ++r)
            bufC[wave][(r + 8 * mh) * 64 + q * 16 + n_] =
                fmaxf(acc[q][r] + bb[q * 16 + n_], 0.0f);
    __syncthreads();
    if (lane < 16) {
        float s = b_b1[0];
        for (int j = 0; j < 64; ++j) s += bufC[wave][lane * 64 + j] * wb1[j];
        baseline[row0 + lane] = s;
    }
}

// ===========================================================================
// K7: per-row argmax over logits (first max, matching jnp.argmax)
// ===========================================================================
__global__ __launch_bounds__(256) void k_argmax(const float* __restrict__ logits,
                                                float* __restrict__ action) {
    int row = blockIdx.x * blockDim.x + threadIdx.x;
    if (row < 2048) {
        const float* p = logits + (long)row * 64;
        float best = p[0];
        int bi = 0;
        for (int j = 1; j < 64; ++j) {
            float v = p[j];
            if (v > best) { best = v; bi = j; }
        }
        action[row] = (float)bi;
    }
}

// ===========================================================================
extern "C" void kernel_launch(void* const* d_in, const int* in_sizes, int n_in,
                              void* d_out, int out_size, void* d_ws, size_t ws_size,
                              hipStream_t stream) {
    const float* state_table   = (const float*)d_in[0];
    const float* state_other   = (const float*)d_in[1];
    const float* actions_table = (const float*)d_in[2];
    const float* actions_other = (const float*)d_in[3];
    const float* h0   = (const float*)d_in[4];
    const float* c0   = (const float*)d_in[5];
    // d_in[6] = offset: uniform K=4 ragged layout (rows 4s..4s+3 per segment)
    const unsigned char* done = (const unsigned char*)d_in[7];
    const float* W_t0 = (const float*)d_in[8];
    const float* b_t0 = (const float*)d_in[9];
    const float* W_t1 = (const float*)d_in[10];
    const float* b_t1 = (const float*)d_in[11];
    const float* W_s0 = (const float*)d_in[12];
    const float* b_s0 = (const float*)d_in[13];
    const float* W_a0 = (const float*)d_in[14];
    const float* b_a0 = (const float*)d_in[15];
    const float* Wih0 = (const float*)d_in[16];
    const float* Whh0 = (const float*)d_in[17];
    const float* bih0 = (const float*)d_in[18];
    const float* bhh0 = (const float*)d_in[19];
    const float* Wih1 = (const float*)d_in[20];
    const float* Whh1 = (const float*)d_in[21];
    const float* bih1 = (const float*)d_in[22];
    const float* bhh1 = (const float*)d_in[23];
    const float* W_p0 = (const float*)d_in[24];
    const float* b_p0 = (const float*)d_in[25];
    const float* W_p1 = (const float*)d_in[26];
    const float* b_p1 = (const float*)d_in[27];
    const float* W_p2 = (const float*)d_in[28];
    const float* b_p2 = (const float*)d_in[29];
    const float* W_p3 = (const float*)d_in[30];
    const float* b_p3 = (const float*)d_in[31];
    const float* W_b0 = (const float*)d_in[32];
    const float* b_b0 = (const float*)d_in[33];
    const float* W_b1 = (const float*)d_in[34];
    const float* b_b1 = (const float*)d_in[35];

    char* ws = (char*)d_ws;
    float*     actions_ws = (float*)ws;                           // 131072*64*4 = 32 MB
    _Float16*  core_in    = (_Float16*)(ws + 33554432);           // 2048*64*2
    float*     state      = (float*)(ws + 33554432 + 262144);     // 2048*64*4

    float* out      = (float*)d_out;
    float* logits   = out;            // 131072
    float* baseline = out + 131072;   // 2048
    float* action   = out + 133120;   // 2048
    float* hT       = out + 135168;   // 4096
    float* cT       = out + 139264;   // 4096

    k_state_tower<<<16, 256, 0, stream>>>(state_table, state_other, W_t0, b_t0,
                                          W_t1, b_t1, W_s0, b_s0, core_in);
    k_feats<<<4096, 256, 0, stream>>>(actions_table, W_t0, b_t0, W_t1, b_t1, actions_ws);
    k_ao<<<1024, 256, 0, stream>>>(actions_other, W_a0, b_a0, actions_ws);
    k_lstm<<<1, 256, 0, stream>>>(core_in, Wih0, Whh0, bih0, bhh0,
                                  Wih1, Whh1, bih1, bhh1, h0, c0, done, state, hT, cT);
    k_pred<<<2048, 128, 0, stream>>>(state, actions_ws, W_p0, b_p0, W_p1, b_p1,
                                     W_p2, b_p2, W_p3, b_p3, logits);
    k_baseline<<<16, 256, 0, stream>>>(state, W_b0, b_b0, W_b1, b_b1, baseline);
    k_argmax<<<8, 256, 0, stream>>>(logits, action);
}
